// MoEManage_25872882991978
// MI455X (gfx1250) — matrix operand order repaired and checked by reading, not compiled, not run on hardware
//
#include <hip/hip_runtime.h>

typedef __attribute__((ext_vector_type(16))) _Float16 v16h;
typedef __attribute__((ext_vector_type(8)))  _Float16 v8h;
typedef __attribute__((ext_vector_type(8)))  float    v8f;

#define B_TOTAL      65536
#define D_IN         2048
#define D_H          256
#define N_EXP        64
#define WAVES_PER_WG 4
#define ROWS_PER_WG  64   // 4 waves * 16 rows

// ---------------------------------------------------------------------------
// Pre-pass: swizzle W2 [256 x 64] fp32 into f16 WMMA B-fragment layout in ws.
// Fragment (kb, t2, lane, e): K = kb*32 + (lane&16) + e, N = t2*16 + (lane&15)
// Linear: o = ((kb*4 + t2)*32 + lane)*16 + e   (16384 halves = 32 KB)
// ---------------------------------------------------------------------------
__global__ __launch_bounds__(256)
void w2_pack(const float* __restrict__ W2, _Float16* __restrict__ w2f)
{
    int o    = blockIdx.x * 256 + threadIdx.x;   // 0..16383
    int e    = o & 15;
    int lane = (o >> 4) & 31;
    int t2   = (o >> 9) & 3;
    int kb   = o >> 11;
    int K    = kb * 32 + (lane & 16) + e;
    int N    = t2 * 16 + (lane & 15);
    w2f[o] = (_Float16)W2[(size_t)K * N_EXP + N];
}

// Load one thread's share of a W1 K-chunk (2 columns x 32 k) into registers,
// converted to f16. Coalesced: consecutive lanes touch consecutive columns.
__device__ __forceinline__ void stage_load(const float* __restrict__ W1,
                                           int kk, int n0, int n1, v8h st[8])
{
    #pragma unroll
    for (int which = 0; which < 2; ++which) {
        const int n = which ? n1 : n0;
        const float* wp = W1 + (size_t)kk * D_H + n;
        #pragma unroll
        for (int g = 0; g < 4; ++g) {
            v8h r;
            #pragma unroll
            for (int k = 0; k < 8; ++k)
                r[k] = (_Float16)wp[(size_t)(g * 8 + k) * D_H];
            st[which * 4 + g] = r;
        }
    }
}

// Store staged registers into LDS in WMMA B-fragment layout:
// (k,n) -> sW1[(tile(n/16)*32 + lane((n&15)|(k&16)))*16 + (k&15)]
__device__ __forceinline__ void stage_store(_Float16* __restrict__ sW1,
                                            int n0, int n1, const v8h st[8])
{
    #pragma unroll
    for (int which = 0; which < 2; ++which) {
        const int n  = which ? n1 : n0;
        const int t  = n >> 4;
        const int lf = n & 15;
        _Float16* d0 = &sW1[(t * 32 + lf) * 16];        // K = 0..15 (e=k)
        _Float16* d1 = &sW1[(t * 32 + lf + 16) * 16];   // K = 16..31 (e=k-16)
        *(v8h*)(d0)     = st[which * 4 + 0];
        *(v8h*)(d0 + 8) = st[which * 4 + 1];
        *(v8h*)(d1)     = st[which * 4 + 2];
        *(v8h*)(d1 + 8) = st[which * 4 + 3];
    }
}

// ---------------------------------------------------------------------------
// Fused router: GEMM1 (2048->256, relu) -> GEMM2 (256->64) -> softmax/top-8
// ---------------------------------------------------------------------------
__global__ __launch_bounds__(128)
void moe_route_fused(const float* __restrict__ tokens,
                     const float* __restrict__ W1,
                     const float* __restrict__ b1,
                     const _Float16* __restrict__ w2f,
                     const float* __restrict__ b2,
                     float* __restrict__ Rout,
                     int*   __restrict__ idxOut)
{
    __shared__ __align__(32) _Float16 sW1[16 * 32 * 16];                    // 16 KB
    // per-wave union: h [16x256] f16 (8KB) | logits [16x64] f32 (4KB)
    __shared__ __align__(32) unsigned char sHL[WAVES_PER_WG][16 * 256 * 2]; // 32 KB

    const int tid   = threadIdx.x;
    const int lane  = tid & 31;
    const int wv    = tid >> 5;
    const int laneM = lane & 15;
    const int hi    = lane >> 4;   // lane half: 0 or 1
    const int koff  = hi * 16;     // A-fragment K offset for this lane half
    const int cM    = hi * 8;      // C-layout row offset for this lane half
    const int rowBase = blockIdx.x * ROWS_PER_WG + wv * 16;

    _Float16* sH = (_Float16*)sHL[wv];
    float*    sL = (float*)sHL[wv];

    const int n0 = tid;           // staging column 0..127
    const int n1 = tid + 128;     // staging column 128..255

    // ---------------- GEMM1: h = relu(x @ W1 + b1), 16 N-tiles per wave ----
    v8f acc[16] = {};

    v8h st[8];
    stage_load(W1, 0, n0, n1, st);          // prologue: chunk 0 -> registers

    for (int kk = 0; kk < D_IN; kk += 32) {
        stage_store(sW1, n0, n1, st);       // registers -> LDS fragments
        __syncthreads();

        const bool more = (kk + 32) < D_IN;
        v8h nx[8];
        if (more)
            stage_load(W1, kk + 32, n0, n1, nx);  // hide under WMMAs below

        // A fragment (16x32 f16): lanes 0-15 K={0..15}, lanes 16-31 K={16..31}
        const float* ap = tokens + (size_t)(rowBase + laneM) * D_IN + kk + koff;
        v16h a;
        #pragma unroll
        for (int e = 0; e < 8; ++e) {
            a[e]     = (_Float16)ap[e];
            a[8 + e] = (_Float16)ap[8 + e];
        }

        // 16 WMMAs with B fragments pipelined one tile ahead
        v16h bcur = *(const v16h*)&sW1[(0 * 32 + lane) << 4];
        #pragma unroll
        for (int t = 0; t < 16; ++t) {
            v16h bnext = bcur;
            if (t < 15)
                bnext = *(const v16h*)&sW1[((t + 1) * 32 + lane) << 4];
            acc[t] = __builtin_amdgcn_wmma_f32_16x16x32_f16(
                false, a, false, bcur, (short)0, acc[t], false, false);
            bcur = bnext;
        }
        __syncthreads();

        if (more) {
            #pragma unroll
            for (int i = 0; i < 8; ++i) st[i] = nx[i];
        }
    }

    // bias + relu; write h to LDS row-major f16 (C layout: M = r+cM, N = t*16+laneM)
    #pragma unroll
    for (int t = 0; t < 16; ++t) {
        float bias = b1[t * 16 + laneM];
        #pragma unroll
        for (int r = 0; r < 8; ++r) {
            float h = acc[t][r] + bias;
            h = h > 0.0f ? h : 0.0f;
            sH[(r + cM) * D_H + t * 16 + laneM] = (_Float16)h;
        }
    }
    __syncthreads();

    // ---------------- GEMM2: logits = h @ W2 + b2, 4 N-tiles per wave ------
    v8f acc2[4] = {};
    #pragma unroll
    for (int kb = 0; kb < 8; ++kb) {
        const int kk = kb * 32;
        const _Float16* hp = sH + laneM * D_H + kk + koff;
        v16h a2;
        #pragma unroll
        for (int e = 0; e < 8; ++e) {
            a2[e]     = hp[e];
            a2[8 + e] = hp[8 + e];
        }
        #pragma unroll
        for (int t2 = 0; t2 < 4; ++t2) {
            v16h bf = *(const v16h*)(w2f + (size_t)(((kb * 4) + t2) * 32 + lane) * 16);
            acc2[t2] = __builtin_amdgcn_wmma_f32_16x16x32_f16(
                false, a2, false, bf, (short)0, acc2[t2], false, false);
        }
    }
    __syncthreads();

    // logits (fp32) into LDS, reusing the h region
    #pragma unroll
    for (int t2 = 0; t2 < 4; ++t2) {
        float bias = b2[t2 * 16 + laneM];
        #pragma unroll
        for (int r = 0; r < 8; ++r)
            sL[(r + cM) * N_EXP + t2 * 16 + laneM] = acc2[t2][r] + bias;
    }
    __syncthreads();

    // ---------------- softmax + top-8 scatter: one row per lane ------------
    if (lane < 16) {
        const int row = rowBase + lane;

        float v[64];
        float mx = -3.402823466e38f;
        #pragma unroll
        for (int i = 0; i < 64; ++i) {
            v[i] = sL[lane * 64 + i];   // TEMPERATURE == 1.0
            mx = fmaxf(mx, v[i]);
        }
        float s = 0.0f;
        #pragma unroll
        for (int i = 0; i < 64; ++i) {
            v[i] = __expf(v[i] - mx);
            s += v[i];
        }
        const float inv = 1.0f / s;

        // 8 selection passes; strict '>' ascending scan => ties pick lowest index
        unsigned long long mask = 0ull;
        int tk[8];
        #pragma unroll
        for (int p = 0; p < 8; ++p) {
            float best = -1.0f;
            int   bi   = 0;
            #pragma unroll
            for (int j = 0; j < 64; ++j) {
                bool freej  = ((mask >> j) & 1ull) == 0ull;
                bool better = freej && (v[j] > best);
                best = better ? v[j] : best;
                bi   = better ? j    : bi;
            }
            mask |= (1ull << bi);
            tk[p] = bi;
        }

        float* rp = Rout + (size_t)row * N_EXP;
        #pragma unroll
        for (int q = 0; q < 16; ++q) {
            float4 o;
            o.x = ((mask >> (4 * q + 0)) & 1ull) ? v[4 * q + 0] * inv : 0.0f;
            o.y = ((mask >> (4 * q + 1)) & 1ull) ? v[4 * q + 1] * inv : 0.0f;
            o.z = ((mask >> (4 * q + 2)) & 1ull) ? v[4 * q + 2] * inv : 0.0f;
            o.w = ((mask >> (4 * q + 3)) & 1ull) ? v[4 * q + 3] * inv : 0.0f;
            ((float4*)rp)[q] = o;
        }

        int4* ip = (int4*)(idxOut + (size_t)row * 8);
        ip[0] = make_int4(tk[0], tk[1], tk[2], tk[3]);
        ip[1] = make_int4(tk[4], tk[5], tk[6], tk[7]);
    }
}

extern "C" void kernel_launch(void* const* d_in, const int* in_sizes, int n_in,
                              void* d_out, int out_size, void* d_ws, size_t ws_size,
                              hipStream_t stream) {
    (void)in_sizes; (void)n_in; (void)ws_size; (void)out_size;

    const float* tokens = (const float*)d_in[0];
    const float* W1     = (const float*)d_in[1];
    const float* b1     = (const float*)d_in[2];
    const float* W2     = (const float*)d_in[3];
    const float* b2     = (const float*)d_in[4];

    float* R   = (float*)d_out;                                   // [B, 64] fp32
    int*   idx = (int*)((float*)d_out + (size_t)B_TOTAL * N_EXP); // [B, 8] int32

    _Float16* w2f = (_Float16*)d_ws;                              // 32 KB fragments

    hipLaunchKernelGGL(w2_pack, dim3(64), dim3(256), 0, stream, W2, w2f);

    dim3 grid(B_TOTAL / ROWS_PER_WG);  // 1024
    dim3 block(128);                   // 4 wave32
    hipLaunchKernelGGL(moe_route_fused, grid, block, 0, stream,
                       tokens, W1, b1, w2f, b2, R, idx);
}
